// GIN_91250875171157
// MI455X (gfx1250) — matrix operand order repaired and checked by reading, not compiled, not run on hardware
//
#include <hip/hip_runtime.h>

// CDNA5 (gfx1250) wave32 WMMA fragment types
typedef __attribute__((ext_vector_type(16))) _Float16 v16h;
typedef __attribute__((ext_vector_type(8)))  float    v8f;

// ---------------------------------------------------------------------------
// Simple vectorized f32 copy (init aggregation buffer with self term, eps=0)
// ---------------------------------------------------------------------------
__global__ void copy_f32(const float4* __restrict__ in, float4* __restrict__ out, long n4) {
    long i = (long)blockIdx.x * blockDim.x + threadIdx.x;
    if (i < n4) out[i] = in[i];
}

// ---------------------------------------------------------------------------
// Edge scatter-add: agg[dst[e], :] += x[src[e], :]
// One thread handles 4 consecutive features -> coalesced 512B reads per edge,
// f32 atomics resolved in L2 (agg buffers are L2-resident: <= 51 MB << 192 MB)
// ---------------------------------------------------------------------------
__global__ void scatter_add(const float* __restrict__ x, const int* __restrict__ src,
                            const int* __restrict__ dst, float* __restrict__ agg,
                            int E, int d) {
    const int per = d >> 2;                    // float4 chunks per row
    long t = (long)blockIdx.x * blockDim.x + threadIdx.x;
    long total = (long)E * per;
    if (t >= total) return;
    int e = (int)(t / per);
    int f = ((int)(t % per)) << 2;
    const float* xs = x   + (size_t)src[e] * d + f;
    float*       ap = agg + (size_t)dst[e] * d + f;
    atomicAdd(ap + 0, xs[0]);
    atomicAdd(ap + 1, xs[1]);
    atomicAdd(ap + 2, xs[2]);
    atomicAdd(ap + 3, xs[3]);
}

// ---------------------------------------------------------------------------
// Fused GEMM + bias (+ReLU):  C[N,M] = act(A[N,K] @ W[K,M] + b[M])
//
// - One wave per 16-row slab; the 16xK A slab is loaded once (float4 /
//   global_load_b128), converted f32->f16 into registers in WMMA A-fragment
//   layout, and reused across all column tiles.
// - Each workgroup owns a 64-column weight panel (blockIdx.y). The panel is
//   cooperatively staged in LDS as f16 in B-fragment order: a lane's whole
//   16-element fragment is 32 contiguous bytes -> one v16h LDS read
//   (ds_load_b128 pair) per WMMA instead of 16 strided global_load_b32.
// - f16 inputs / f32 accumulation via v_wmma_f32_16x16x32_f16.
// ---------------------------------------------------------------------------
template<int K, bool RELU>
__global__ __launch_bounds__(256) void gemm_wmma(const float* __restrict__ A,
                                                 const float* __restrict__ W,
                                                 const float* __restrict__ bias,
                                                 float* __restrict__ C,
                                                 int N, int M) {
    constexpr int KS = K / 32;                 // wmma K-steps
    // B panel in fragment order: block (s*2 + h)*64 + c  holds 16 f16 for
    // K = s*32 + h*16 + 0..15 of panel column c.  Size: K*64 f16 (<= 32 KB).
    __shared__ __align__(32) _Float16 Bsh[KS * 2 * 64 * 16];

    const int ct0 = (int)blockIdx.y * 64;      // panel start column (M % 64 == 0)

    // ---- cooperative panel fill: coalesced f32 reads, f16 LDS stores ----
    for (int idx = threadIdx.x; idx < K * 64; idx += 256) {
        int k = idx >> 6;                      // 0..K-1
        int c = idx & 63;                      // panel-local column
        float w = W[(size_t)k * M + (ct0 + c)];
        int s = k >> 5, h = (k >> 4) & 1, r = k & 15;
        Bsh[((((s * 2 + h) * 64) + c) << 4) + r] = (_Float16)w;
    }
    __syncthreads();

    const int lane = threadIdx.x & 31;
    const int wid  = (int)(blockIdx.x * (blockDim.x >> 5)) + (threadIdx.x >> 5);
    const int row0 = wid * 16;
    if (row0 + 16 > N) return;                 // wave-uniform; after barrier

    const int mrow = lane & 15;                // A row / B,C column within tile
    const int half = lane >> 4;                // lane half selects K/M sub-block

    // ---- load 16 x K A-slab, convert f32 -> f16 into A-fragment layout ----
    // Lane's group-of-4 pairs are 8 consecutive floats -> two float4 loads.
    v16h afrag[KS];
    const float* arow = A + (size_t)(row0 + mrow) * K;
#pragma unroll
    for (int s = 0; s < KS; ++s) {
#pragma unroll
        for (int g = 0; g < 2; ++g) {          // fragment halves (K base 0 / 8)
            const float* p = arow + s * 32 + half * 16 + g * 8;
            float4 a = *(const float4*)(p + 0);
            float4 b = *(const float4*)(p + 4);
            afrag[s][g * 8 + 0] = (_Float16)a.x;
            afrag[s][g * 8 + 1] = (_Float16)a.y;
            afrag[s][g * 8 + 2] = (_Float16)a.z;
            afrag[s][g * 8 + 3] = (_Float16)a.w;
            afrag[s][g * 8 + 4] = (_Float16)b.x;
            afrag[s][g * 8 + 5] = (_Float16)b.y;
            afrag[s][g * 8 + 6] = (_Float16)b.z;
            afrag[s][g * 8 + 7] = (_Float16)b.w;
        }
    }

#pragma unroll
    for (int tl = 0; tl < 4; ++tl) {           // 4 col tiles per 64-col panel
        const int colLocal = tl * 16 + mrow;
        const int ncol     = ct0 + colLocal;
        v8f acc = {};
#pragma unroll
        for (int s = 0; s < KS; ++s) {
            // whole B fragment = 32 contiguous, 32B-aligned bytes in LDS
            const v16h bfrag =
                *(const v16h*)(Bsh + ((((s * 2 + half) * 64) + colLocal) << 4));
            acc = __builtin_amdgcn_wmma_f32_16x16x32_f16(
                /*neg_a=*/false, afrag[s], /*neg_b=*/false, bfrag,
                /*c_mod=*/(short)0, acc, /*reuse_a=*/false, /*reuse_b=*/false);
        }
        // C/D 16x16 f32: VGPR i -> row i + half*8, lane%16 -> col
        const float bv = bias[ncol];
#pragma unroll
        for (int i = 0; i < 8; ++i) {
            float v = acc[i] + bv;
            if (RELU) v = fmaxf(v, 0.0f);
            C[(size_t)(row0 + half * 8 + i) * M + ncol] = v;
        }
    }
}

// ---------------------------------------------------------------------------
// Pipeline:
//  agg0 = x ; agg0 += scatter(x)                 (bufA, N x 128)
//  t1   = relu(agg0 @ W1a + b1a)                 (bufB, N x 256)
//  h1   = relu(t1   @ W2a + b2a)                 (bufC, N x 256)  [layer-0 out]
//  agg1 = h1 ; agg1 += scatter(h1)               (bufA reused, N x 256)
//  t2   = relu(agg1 @ W1b + b1b)                 (bufB reused, N x 256)
//  out  =      t2   @ W2b + b2b                  (d_out, N x 128)
// ---------------------------------------------------------------------------
extern "C" void kernel_launch(void* const* d_in, const int* in_sizes, int n_in,
                              void* d_out, int out_size, void* d_ws, size_t ws_size,
                              hipStream_t stream) {
    const float* x   = (const float*)d_in[0];
    const float* W1a = (const float*)d_in[1];
    const float* b1a = (const float*)d_in[2];
    const float* W2a = (const float*)d_in[3];
    const float* b2a = (const float*)d_in[4];
    const float* W1b = (const float*)d_in[5];
    const float* b1b = (const float*)d_in[6];
    const float* W2b = (const float*)d_in[7];
    const float* b2b = (const float*)d_in[8];
    const int*   ei  = (const int*)d_in[9];    // edge_index flat [2,E] int32

    const int DI = 128, DH = 256, DO = 128;
    const int N = in_sizes[0] / DI;
    const int E = in_sizes[9] / 2;
    const int* srcp = ei;
    const int* dstp = ei + E;

    float* bufA = (float*)d_ws;                 // N*256 f32
    float* bufB = bufA + (size_t)N * DH;        // N*256 f32
    float* bufC = bufB + (size_t)N * DH;        // N*256 f32

    dim3 blk(256);
    const unsigned gx = (unsigned)((N / 16 + 7) / 8);   // 8 waves per block

    // layer 0 aggregation: bufA = x + sum_{src->dst} x[src]
    {
        long n4 = (long)N * DI / 4;
        copy_f32<<<dim3((unsigned)((n4 + 255) / 256)), blk, 0, stream>>>(
            (const float4*)x, (float4*)bufA, n4);
        long tot = (long)E * (DI / 4);
        scatter_add<<<dim3((unsigned)((tot + 255) / 256)), blk, 0, stream>>>(
            x, srcp, dstp, bufA, E, DI);
    }
    // layer 0 MLP + outer ReLU
    gemm_wmma<128, true><<<dim3(gx, DH / 64), blk, 0, stream>>>(bufA, W1a, b1a, bufB, N, DH);
    gemm_wmma<256, true><<<dim3(gx, DH / 64), blk, 0, stream>>>(bufB, W2a, b2a, bufC, N, DH);

    // layer 1 aggregation: bufA = h1 + sum h1[src]
    {
        long n4 = (long)N * DH / 4;
        copy_f32<<<dim3((unsigned)((n4 + 255) / 256)), blk, 0, stream>>>(
            (const float4*)bufC, (float4*)bufA, n4);
        long tot = (long)E * (DH / 4);
        scatter_add<<<dim3((unsigned)((tot + 255) / 256)), blk, 0, stream>>>(
            bufC, srcp, dstp, bufA, E, DH);
    }
    // layer 1 MLP (no trailing activation)
    gemm_wmma<256, true ><<<dim3(gx, DH / 64), blk, 0, stream>>>(bufA, W1b, b1b, bufB, N, DH);
    gemm_wmma<256, false><<<dim3(gx, DO / 64), blk, 0, stream>>>(bufB, W2b, b2b, (float*)d_out, N, DO);
}
